// Fp8MoEMethod_73100343378288
// MI455X (gfx1250) — matrix operand order repaired and checked by reading, not compiled, not run on hardware
//
#include <hip/hip_runtime.h>
#include <hip/hip_bf16.h>

// Problem constants (match reference)
#define E_  8
#define T_  512
#define H_  2048
#define I_  1024

typedef __bf16 bf16_t;
typedef __attribute__((ext_vector_type(16))) __bf16 v16bf;
typedef __attribute__((ext_vector_type(8)))  __bf16 bf16x8;
typedef __attribute__((ext_vector_type(8)))  float  v8f;
typedef unsigned int u32x4 __attribute__((ext_vector_type(4)));
typedef int          i32x8 __attribute__((ext_vector_type(8)));
typedef int          i32x4 __attribute__((ext_vector_type(4)));

// LDS tile: 64 rows x 128 dwords, padded 4 dwords per row -> pitch 132 dwords
// (TDM pad_interval=6 (128 dw), pad_amount=3 (4 dw)); kills all bank conflicts
// for the 16B ds_load_b128 B-fragment reads (528 B row pitch -> 16 B bank skew).
#define TILE_PITCH 132
#define TILE_DWORDS (64 * TILE_PITCH)

#if defined(__HIP_DEVICE_COMPILE__) && __has_builtin(__builtin_amdgcn_tensor_load_to_lds)
#define USE_TDM 1
#else
#define USE_TDM 0
#endif

#if defined(__has_include)
#  if __has_include(<hip/amd_detail/amd_gfx1250_TDM.h>)
#    define TDM_6ARG 1
#  else
#    define TDM_6ARG 0
#  endif
#else
#  define TDM_6ARG 0
#endif

#if defined(__HIP_DEVICE_COMPILE__)
#  if __has_builtin(__builtin_amdgcn_s_wait_tensorcnt)
#    define WAIT_TENSOR(n) __builtin_amdgcn_s_wait_tensorcnt(n)
#  else
#    define WAIT_TENSOR(n) asm volatile("s_wait_tensorcnt %0" :: "i"(n) : "memory")
#  endif
#else
#  define WAIT_TENSOR(n) ((void)0)
#endif

// ---------------------------------------------------------------------------
// TDM: 2D tile load, tile_d1 rows x tile_d0 fp32 elements, global row stride
// `stride0` elements, LDS rows padded to TILE_PITCH dwords by the TDM.
// ---------------------------------------------------------------------------
__device__ __forceinline__ void tdm_load_2d(unsigned lds_addr, const float* gptr,
                                            unsigned tensor_d0, unsigned tensor_d1,
                                            unsigned tile_d0, unsigned tile_d1,
                                            unsigned stride0) {
#if USE_TDM
    unsigned long long ga = (unsigned long long)(uintptr_t)gptr;
    // D# group 0: count=1, lds_addr, global_addr[56:0], type=2 ("image")
    u32x4 g0 = { 1u,
                 lds_addr,
                 (unsigned)ga,
                 (unsigned)(ga >> 32) | (2u << 30) };
    // D# group 1: data_size=2 (4B), pad_enable=1, pad_interval=6 (128 dw),
    // pad_amount=3 (4 dw); dims/strides per ISA Ch.8 bit layout.
    i32x8 g1 = { (int)(0x00020000u | (1u << 20) | (6u << 22) | (3u << 25)),
                 (int)(tensor_d0 << 16),
                 (int)((tensor_d0 >> 16) | (tensor_d1 << 16)),
                 (int)((tensor_d1 >> 16) | (tile_d0 << 16)),
                 (int)tile_d1,
                 (int)stride0,
                 0, 0 };
    i32x4 gz = { 0, 0, 0, 0 };
#if TDM_6ARG
    i32x8 gz8 = { 0, 0, 0, 0, 0, 0, 0, 0 };
    __builtin_amdgcn_tensor_load_to_lds(g0, g1, gz, gz, gz8, 0);
#else
    __builtin_amdgcn_tensor_load_to_lds(g0, g1, gz, gz, 0);
#endif
#else
    (void)lds_addr; (void)gptr; (void)tensor_d0; (void)tensor_d1;
    (void)tile_d0; (void)tile_d1; (void)stride0;
#endif
}

#if !USE_TDM
// Fallback staging: cooperative 64x128 fp32 copy into padded LDS tile.
__device__ __forceinline__ void stage_tile_fallback(float* sh, const float* g,
                                                    int stride0, int tid) {
    for (int idx = tid; idx < 64 * 32; idx += 128) {
        int r = idx >> 5, c4 = (idx & 31) * 4;
        *(float4*)(sh + r * TILE_PITCH + c4) =
            *(const float4*)(g + (size_t)r * stride0 + c4);
    }
}
#endif

// ---------------------------------------------------------------------------
// Fragment loaders (wave32 WMMA; A and B both use contiguous k = 16*(lane/16) + s)
// ---------------------------------------------------------------------------

// A 16x32 bf16 fragment: lane holds row M = lane&15.
// lanes 0-15:  K = k0+0..15 ; lanes 16-31: K = k0+16..31
__device__ __forceinline__ v16bf load_a_frag(const bf16_t* __restrict__ arow_k0,
                                             int hf) {
    bf16x8 c0 = *(const bf16x8*)(arow_k0 + hf * 16);
    bf16x8 c1 = *(const bf16x8*)(arow_k0 + hf * 16 + 8);
    v16bf a;
#pragma unroll
    for (int i = 0; i < 8; ++i) { a[i] = c0[i]; a[8 + i] = c1[i]; }
    return a;
}

// B 32x16 bf16 fragment from an fp32 row (LDS tile): lane = column N.
// Loads all four float4 first (MLP), then dequant+cvt to bf16.
__device__ __forceinline__ v16bf cvt_b_frag(const float* p, float s) {
    float4 w0 = ((const float4*)p)[0];
    float4 w1 = ((const float4*)p)[1];
    float4 w2 = ((const float4*)p)[2];
    float4 w3 = ((const float4*)p)[3];
    v16bf b;
    b[0]  = (bf16_t)(w0.x * s); b[1]  = (bf16_t)(w0.y * s);
    b[2]  = (bf16_t)(w0.z * s); b[3]  = (bf16_t)(w0.w * s);
    b[4]  = (bf16_t)(w1.x * s); b[5]  = (bf16_t)(w1.y * s);
    b[6]  = (bf16_t)(w1.z * s); b[7]  = (bf16_t)(w1.w * s);
    b[8]  = (bf16_t)(w2.x * s); b[9]  = (bf16_t)(w2.y * s);
    b[10] = (bf16_t)(w2.z * s); b[11] = (bf16_t)(w2.w * s);
    b[12] = (bf16_t)(w3.x * s); b[13] = (bf16_t)(w3.y * s);
    b[14] = (bf16_t)(w3.z * s); b[15] = (bf16_t)(w3.w * s);
    return b;
}

// ---------------------------------------------------------------------------
// Kernel 1: zero output + expert counters
// ---------------------------------------------------------------------------
__global__ void moe_zero_kernel(float* __restrict__ out, int n4,
                                int* __restrict__ cnt) {
    int i = blockIdx.x * blockDim.x + threadIdx.x;
    if (i < n4) ((float4*)out)[i] = make_float4(0.f, 0.f, 0.f, 0.f);
    if (blockIdx.x == 0 && threadIdx.x < E_) cnt[threadIdx.x] = 0;
}

// ---------------------------------------------------------------------------
// Kernel 2: routing — softmax, top-2, renormalize, build expert token lists
// ---------------------------------------------------------------------------
__global__ void moe_routing_kernel(const float* __restrict__ logits,
                                   int* __restrict__ cnt,
                                   int* __restrict__ etok,
                                   float* __restrict__ egw) {
    int t = blockIdx.x * blockDim.x + threadIdx.x;
    if (t >= T_) return;
    float l[E_];
    float m = -1e30f;
#pragma unroll
    for (int e = 0; e < E_; ++e) { l[e] = logits[t * E_ + e]; m = fmaxf(m, l[e]); }
#pragma unroll
    for (int e = 0; e < E_; ++e) l[e] = __expf(l[e] - m);
    int i0 = 0; float v0 = l[0];
#pragma unroll
    for (int e = 1; e < E_; ++e) if (l[e] > v0) { v0 = l[e]; i0 = e; }
    int i1 = (i0 == 0) ? 1 : 0; float v1 = l[i1];
#pragma unroll
    for (int e = 0; e < E_; ++e) if (e != i0 && l[e] > v1) { v1 = l[e]; i1 = e; }
    float inv = 1.f / (v0 + v1);
    int s0 = atomicAdd(&cnt[i0], 1);
    etok[i0 * T_ + s0] = t; egw[i0 * T_ + s0] = v0 * inv;
    int s1 = atomicAdd(&cnt[i1], 1);
    etok[i1 * T_ + s1] = t; egw[i1 * T_ + s1] = v1 * inv;
}

// ---------------------------------------------------------------------------
// Kernel 3: x fp32 -> bf16
// ---------------------------------------------------------------------------
__global__ void moe_cvtx_kernel(const float* __restrict__ x,
                                bf16_t* __restrict__ xb) {
    int i = (blockIdx.x * blockDim.x + threadIdx.x) * 8;
    if (i >= T_ * H_) return;
    float4 a = *(const float4*)(x + i);
    float4 b = *(const float4*)(x + i + 4);
    bf16x8 o;
    o[0] = (bf16_t)a.x; o[1] = (bf16_t)a.y; o[2] = (bf16_t)a.z; o[3] = (bf16_t)a.w;
    o[4] = (bf16_t)b.x; o[5] = (bf16_t)b.y; o[6] = (bf16_t)b.z; o[7] = (bf16_t)b.w;
    *(bf16x8*)(xb + i) = o;
}

// ---------------------------------------------------------------------------
// Kernel 4: GEMM1 (x @ w13^T, block-dequant) fused with silu_and_mul.
// grid = (I/64, T/32 token tiles, E), block = 128 (4 waves).
// Block tile: 32 tokens x 64 cols of BOTH gate and up halves.
// Weights double-buffered via TDM: wave0 issues the NEXT 64x128 fp32 tile
// pair, then waits tensorcnt<=2 (releases current pair, next pair overlaps
// compute). Each B-fragment feeds 2 WMMAs (two 16-token A fragments).
// ---------------------------------------------------------------------------
__global__ void __launch_bounds__(128)
moe_gemm1_silu_kernel(const float* __restrict__ w13,
                      const float* __restrict__ w13s,
                      const bf16_t* __restrict__ xb,
                      const int* __restrict__ cnt,
                      const int* __restrict__ etok,
                      bf16_t* __restrict__ abuf) {
    __shared__ __align__(16) float shg[2][TILE_DWORDS];
    __shared__ __align__(16) float shu[2][TILE_DWORDS];

    const int e = blockIdx.z;
    const int n_tok = cnt[e];
    const int ttile = blockIdx.y;
    if (ttile * 32 >= n_tok) return;

    const int wave = threadIdx.x >> 5, lane = threadIdx.x & 31;
    const int hf = lane >> 4, l15 = lane & 15;
    const int nb_blk = blockIdx.x * 64;           // block's column base within I
    const int nloc = wave * 16 + l15;             // lane's column within block tile

    // A: two gathered token rows (clamped for tile padding; pad rows never stored)
    int slot0 = ttile * 32 + l15;      if (slot0 >= n_tok) slot0 = n_tok - 1;
    int slot1 = ttile * 32 + 16 + l15; if (slot1 >= n_tok) slot1 = n_tok - 1;
    const bf16_t* arow0 = xb + (size_t)etok[e * T_ + slot0] * H_;
    const bf16_t* arow1 = xb + (size_t)etok[e * T_ + slot1] * H_;

    // Dequant scale rows for this lane's gate/up columns
    const int rg = nb_blk + nloc;
    const int ru = I_ + nb_blk + nloc;
    const float* sgrow = w13s + ((size_t)e * (2 * I_ / 128) + (rg >> 7)) * (H_ / 128);
    const float* surow = w13s + ((size_t)e * (2 * I_ / 128) + (ru >> 7)) * (H_ / 128);

    const float* gbase = w13 + ((size_t)e * 2 * I_ + nb_blk) * H_;
    const float* ubase = w13 + ((size_t)e * 2 * I_ + I_ + nb_blk) * H_;

    const int NKB = H_ / 128;
#if USE_TDM
    if (threadIdx.x < 32) {                        // prologue: stage buffer 0
        tdm_load_2d((unsigned)(uintptr_t)&shg[0][0], gbase, H_, 1u << 20, 128, 64, H_);
        tdm_load_2d((unsigned)(uintptr_t)&shu[0][0], ubase, H_, 1u << 20, 128, 64, H_);
    }
#endif

    v8f cg0 = {}, cg1 = {}, cu0 = {}, cu1 = {};
    for (int i = 0; i < NKB; ++i) {
        const int kb = i * 128;
        const int cur = i & 1;
#if USE_TDM
        if (threadIdx.x < 32) {
            if (i + 1 < NKB) {                     // issue next pair, then release current
                tdm_load_2d((unsigned)(uintptr_t)&shg[cur ^ 1][0],
                            gbase + kb + 128, H_, 1u << 20, 128, 64, H_);
                tdm_load_2d((unsigned)(uintptr_t)&shu[cur ^ 1][0],
                            ubase + kb + 128, H_, 1u << 20, 128, 64, H_);
                WAIT_TENSOR(2);
            } else {
                WAIT_TENSOR(0);
            }
        }
#else
        stage_tile_fallback(&shg[cur][0], gbase + kb, H_, threadIdx.x);
        stage_tile_fallback(&shu[cur][0], ubase + kb, H_, threadIdx.x);
#endif
        __syncthreads();                           // current buffer visible to all
        const float sg = sgrow[kb >> 7];
        const float su = surow[kb >> 7];
#pragma unroll
        for (int kk = 0; kk < 128; kk += 32) {
            v16bf a0 = load_a_frag(arow0 + kb + kk, hf);
            v16bf a1 = load_a_frag(arow1 + kb + kk, hf);
            v16bf bg = cvt_b_frag(&shg[cur][nloc * TILE_PITCH + kk + hf * 16], sg);
            v16bf bu = cvt_b_frag(&shu[cur][nloc * TILE_PITCH + kk + hf * 16], su);
            cg0 = __builtin_amdgcn_wmma_f32_16x16x32_bf16(
                      false, a0, false, bg, (short)0, cg0, false, false);
            cg1 = __builtin_amdgcn_wmma_f32_16x16x32_bf16(
                      false, a1, false, bg, (short)0, cg1, false, false);
            cu0 = __builtin_amdgcn_wmma_f32_16x16x32_bf16(
                      false, a0, false, bu, (short)0, cu0, false, false);
            cu1 = __builtin_amdgcn_wmma_f32_16x16x32_bf16(
                      false, a1, false, bu, (short)0, cu1, false, false);
        }
        __syncthreads();                           // reads done before buffer reuse
    }

    // a = silu(gate) * up, stored bf16 to abuf[e][row][col]
#pragma unroll
    for (int j = 0; j < 8; ++j) {
        const int m = hf * 8 + j;                  // C/D: lane=col N, vgpr j = row M
        const int r0 = ttile * 32 + m;
        const int r1 = ttile * 32 + 16 + m;
        if (r0 < n_tok) {
            const float g = cg0[j];
            const float a = (g / (1.f + __expf(-g))) * cu0[j];
            abuf[((size_t)e * T_ + r0) * I_ + nb_blk + nloc] = (bf16_t)a;
        }
        if (r1 < n_tok) {
            const float g = cg1[j];
            const float a = (g / (1.f + __expf(-g))) * cu1[j];
            abuf[((size_t)e * T_ + r1) * I_ + nb_blk + nloc] = (bf16_t)a;
        }
    }
}

// ---------------------------------------------------------------------------
// Kernel 5: GEMM2 (a @ w2^T, block-dequant) + gated scatter into out.
// grid = (H/64, T/32 token tiles, E), block = 128 (4 waves).
// Double-buffered TDM weight staging (1 tile in flight -> tensorcnt<=1).
// Each token receives exactly 2 atomic contributions -> deterministic sum.
// ---------------------------------------------------------------------------
__global__ void __launch_bounds__(128)
moe_gemm2_scatter_kernel(const float* __restrict__ w2,
                         const float* __restrict__ w2s,
                         const bf16_t* __restrict__ abuf,
                         const int* __restrict__ cnt,
                         const int* __restrict__ etok,
                         const float* __restrict__ egw,
                         float* __restrict__ out) {
    __shared__ __align__(16) float shw[2][TILE_DWORDS];

    const int e = blockIdx.z;
    const int n_tok = cnt[e];
    const int ttile = blockIdx.y;
    if (ttile * 32 >= n_tok) return;

    const int wave = threadIdx.x >> 5, lane = threadIdx.x & 31;
    const int hf = lane >> 4, l15 = lane & 15;
    const int nb_blk = blockIdx.x * 64;            // block's column base within H
    const int nloc = wave * 16 + l15;

    int slot0 = ttile * 32 + l15;      if (slot0 >= n_tok) slot0 = n_tok - 1;
    int slot1 = ttile * 32 + 16 + l15; if (slot1 >= n_tok) slot1 = n_tok - 1;
    const bf16_t* arow0 = abuf + ((size_t)e * T_ + slot0) * I_;
    const bf16_t* arow1 = abuf + ((size_t)e * T_ + slot1) * I_;

    const int hr = nb_blk + nloc;                  // w2 output row (H dim)
    const float* srow = w2s + ((size_t)e * (H_ / 128) + (hr >> 7)) * (I_ / 128);
    const float* wbase = w2 + ((size_t)e * H_ + nb_blk) * I_;

    const int NKB = I_ / 128;
#if USE_TDM
    if (threadIdx.x < 32) {
        tdm_load_2d((unsigned)(uintptr_t)&shw[0][0], wbase, I_, 1u << 20, 128, 64, I_);
    }
#endif

    v8f c0 = {}, c1 = {};
    for (int i = 0; i < NKB; ++i) {
        const int kb = i * 128;
        const int cur = i & 1;
#if USE_TDM
        if (threadIdx.x < 32) {
            if (i + 1 < NKB) {
                tdm_load_2d((unsigned)(uintptr_t)&shw[cur ^ 1][0],
                            wbase + kb + 128, I_, 1u << 20, 128, 64, I_);
                WAIT_TENSOR(1);
            } else {
                WAIT_TENSOR(0);
            }
        }
#else
        stage_tile_fallback(&shw[cur][0], wbase + kb, I_, threadIdx.x);
#endif
        __syncthreads();
        const float s = srow[kb >> 7];
#pragma unroll
        for (int kk = 0; kk < 128; kk += 32) {
            v16bf a0 = load_a_frag(arow0 + kb + kk, hf);
            v16bf a1 = load_a_frag(arow1 + kb + kk, hf);
            v16bf bw = cvt_b_frag(&shw[cur][nloc * TILE_PITCH + kk + hf * 16], s);
            c0 = __builtin_amdgcn_wmma_f32_16x16x32_bf16(
                     false, a0, false, bw, (short)0, c0, false, false);
            c1 = __builtin_amdgcn_wmma_f32_16x16x32_bf16(
                     false, a1, false, bw, (short)0, c1, false, false);
        }
        __syncthreads();
    }

#pragma unroll
    for (int j = 0; j < 8; ++j) {
        const int m = hf * 8 + j;
        const int r0 = ttile * 32 + m;
        const int r1 = ttile * 32 + 16 + m;
        if (r0 < n_tok) {
            atomicAdd(&out[(size_t)etok[e * T_ + r0] * H_ + nb_blk + nloc],
                      egw[e * T_ + r0] * c0[j]);
        }
        if (r1 < n_tok) {
            atomicAdd(&out[(size_t)etok[e * T_ + r1] * H_ + nb_blk + nloc],
                      egw[e * T_ + r1] * c1[j]);
        }
    }
}

// ---------------------------------------------------------------------------
// Launch
// ---------------------------------------------------------------------------
extern "C" void kernel_launch(void* const* d_in, const int* in_sizes, int n_in,
                              void* d_out, int out_size, void* d_ws, size_t ws_size,
                              hipStream_t stream) {
    const float* x      = (const float*)d_in[0];   // [T, H]
    const float* logits = (const float*)d_in[1];   // [T, E]
    const float* w13    = (const float*)d_in[2];   // [E, 2I, H]
    const float* w13s   = (const float*)d_in[3];   // [E, 2I/128, H/128]
    const float* w2     = (const float*)d_in[4];   // [E, H, I]
    const float* w2s    = (const float*)d_in[5];   // [E, H/128, I/128]
    float* out = (float*)d_out;                    // [T, H]

    // Workspace layout (256B aligned)
    char* ws = (char*)d_ws;
    bf16_t* xb   = (bf16_t*)(ws);                              // T*H*2      = 2 MiB
    bf16_t* abuf = (bf16_t*)(ws + 2097152);                    // E*T*I*2    = 8 MiB
    int*    cnt  = (int*)   (ws + 2097152 + 8388608);          // E*4
    int*    etok = (int*)   (ws + 2097152 + 8388608 + 256);    // E*T*4
    float*  egw  = (float*) (ws + 2097152 + 8388608 + 256 + 16384);
    (void)in_sizes; (void)n_in; (void)out_size; (void)ws_size;

    moe_zero_kernel<<<(T_ * H_ / 4 + 255) / 256, 256, 0, stream>>>(out, T_ * H_ / 4, cnt);
    moe_routing_kernel<<<(T_ + 255) / 256, 256, 0, stream>>>(logits, cnt, etok, egw);
    moe_cvtx_kernel<<<(T_ * H_ / 8 + 255) / 256, 256, 0, stream>>>(x, xb);
    moe_gemm1_silu_kernel<<<dim3(I_ / 64, T_ / 32, E_), 128, 0, stream>>>(
        w13, w13s, xb, cnt, etok, abuf);
    moe_gemm2_scatter_kernel<<<dim3(H_ / 64, T_ / 32, E_), 128, 0, stream>>>(
        w2, w2s, abuf, cnt, etok, egw, out);
}
